// BigBirdForLEDGAR_29111288333017
// MI455X (gfx1250) — matrix operand order repaired and checked by reading, not compiled, not run on hardware
//
#include <hip/hip_runtime.h>
#include <hip/hip_bf16.h>
#include <math.h>

// ---------------- model constants ----------------
#define B_    2
#define S_    4096
#define D_    768
#define H_    12
#define HD_   64
#define L_    12
#define FFN_  3072
#define BLK_  64
#define NSEL_ 8
#define NBLK_ 64
#define NLAB_ 100
#define SCALE_ (1.0f/8.0f)

typedef __bf16 bf16_t;
typedef __attribute__((ext_vector_type(16))) __bf16 v16bf;
typedef __attribute__((ext_vector_type(8)))  __bf16 v8bf;
typedef __attribute__((ext_vector_type(8)))  float  v8f;

// A-matrix fragment (16x32 bf16): lane half `hi` holds 16 contiguous K starting at 16*hi
__device__ __forceinline__ v16bf ld_fragA(const bf16_t* p, int hi) {
  union { v16bf v; v8bf h[2]; } u;
  u.h[0] = *reinterpret_cast<const v8bf*>(p + 16*hi);
  u.h[1] = *reinterpret_cast<const v8bf*>(p + 16*hi + 8);
  return u.v;
}
// B-matrix fragment (32x16 bf16): lane half `hi` holds 16 contiguous K starting at 16*hi (column = lane%16)
__device__ __forceinline__ v16bf ld_fragB(const bf16_t* p, int hi) {
  union { v16bf v; v8bf h[2]; } u;
  u.h[0] = *reinterpret_cast<const v8bf*>(p + 16*hi);
  u.h[1] = *reinterpret_cast<const v8bf*>(p + 16*hi + 8);
  return u.v;
}
__device__ __forceinline__ v8f wmma_bf16(v16bf a, v16bf b, v8f c) {
  return __builtin_amdgcn_wmma_f32_16x16x32_bf16(false, a, false, b, (short)0, c, false, false);
}
__device__ __forceinline__ v8f vzero8() {
  v8f z;
  #pragma unroll
  for (int i = 0; i < 8; ++i) z[i] = 0.0f;
  return z;
}

// ---------------- generic WMMA GEMM: C[M,N] = epi(A[M,K] @ B[K,N] + bias (+R)) ----------------
#define BM 128
#define BN 64
#define BK 32
#define LDA_P 40   // 80B row stride (16B multiple)
#define LDB_P 40

// EPI: 0 = bias only, 1 = gelu_new(bias+..), 2 = bias + residual R
// Double-buffered LDS stages: global->regs prefetch of stage kt+1 overlaps WMMA on stage kt;
// one barrier per K-step. fp32->bf16 conversion folded into the LDS store.
template <int EPI>
__global__ __launch_bounds__(256)
void gemm_bf16_wmma(const float* __restrict__ A, const float* __restrict__ Bm,
                    const float* __restrict__ bias, const float* __restrict__ R,
                    float* __restrict__ C, int M, int N, int K)
{
  __shared__ bf16_t As[2][BM][LDA_P];
  __shared__ bf16_t Bs[2][BN][LDB_P];

  const int tid  = threadIdx.x;
  const int lane = tid & 31;
  const int wave = tid >> 5;
  const int wm   = wave & 3;   // 4 waves along M (32 rows each)
  const int wn   = wave >> 2;  // 2 waves along N (32 cols each)
  const int lo   = lane & 15;
  const int hi   = lane >> 4;
  const int m0   = blockIdx.y * BM;
  const int n0   = blockIdx.x * BN;

  v8f acc[2][2];
  acc[0][0] = vzero8(); acc[0][1] = vzero8();
  acc[1][0] = vzero8(); acc[1][1] = vzero8();

  float ar[16];  // staged A (128x32 tile / 256 threads)
  float br[8];   // staged B (32x64 tile / 256 threads)

  // prologue: stage kt=0
  #pragma unroll
  for (int i = 0; i < 16; ++i) {
    int idx = i * 256 + tid;
    ar[i] = A[(size_t)(m0 + (idx >> 5)) * K + (idx & 31)];
  }
  #pragma unroll
  for (int i = 0; i < 8; ++i) {
    int idx = i * 256 + tid;
    br[i] = Bm[(size_t)(idx >> 6) * N + n0 + (idx & 63)];
  }
  #pragma unroll
  for (int i = 0; i < 16; ++i) {
    int idx = i * 256 + tid;
    As[0][idx >> 5][idx & 31] = (bf16_t)ar[i];
  }
  #pragma unroll
  for (int i = 0; i < 8; ++i) {
    int idx = i * 256 + tid;
    Bs[0][idx & 63][idx >> 6] = (bf16_t)br[i];
  }
  __syncthreads();

  const int nk = K / BK;
  for (int kt = 0; kt < nk; ++kt) {
    const int cur = kt & 1;
    if (kt + 1 < nk) {
      const int kb = (kt + 1) * BK;
      // speculative prefetch two stages ahead (global_prefetch_b8)
      if (kt + 2 < nk) {
        __builtin_prefetch(&A[(size_t)(m0 + (tid >> 3)) * K + (kt + 2) * BK], 0, 1);
        __builtin_prefetch(&Bm[(size_t)((kt + 2) * BK + (tid >> 6)) * N + n0 + (tid & 63)], 0, 1);
      }
      #pragma unroll
      for (int i = 0; i < 16; ++i) {
        int idx = i * 256 + tid;
        ar[i] = A[(size_t)(m0 + (idx >> 5)) * K + kb + (idx & 31)];
      }
      #pragma unroll
      for (int i = 0; i < 8; ++i) {
        int idx = i * 256 + tid;
        br[i] = Bm[(size_t)(kb + (idx >> 6)) * N + n0 + (idx & 63)];
      }
    }

    v16bf a0 = ld_fragA(&As[cur][wm * 32 +      lo][0], hi);
    v16bf a1 = ld_fragA(&As[cur][wm * 32 + 16 + lo][0], hi);
    v16bf b0 = ld_fragB(&Bs[cur][wn * 32 +      lo][0], hi);
    v16bf b1 = ld_fragB(&Bs[cur][wn * 32 + 16 + lo][0], hi);

    acc[0][0] = wmma_bf16(a0, b0, acc[0][0]);
    acc[0][1] = wmma_bf16(a0, b1, acc[0][1]);
    acc[1][0] = wmma_bf16(a1, b0, acc[1][0]);
    acc[1][1] = wmma_bf16(a1, b1, acc[1][1]);

    if (kt + 1 < nk) {
      #pragma unroll
      for (int i = 0; i < 16; ++i) {
        int idx = i * 256 + tid;
        As[cur ^ 1][idx >> 5][idx & 31] = (bf16_t)ar[i];
      }
      #pragma unroll
      for (int i = 0; i < 8; ++i) {
        int idx = i * 256 + tid;
        Bs[cur ^ 1][idx & 63][idx >> 6] = (bf16_t)br[i];
      }
    }
    __syncthreads();
  }

  // epilogue: C frag layout: VGPR r, lane -> (M = r + 8*hi, N = lo) within 16x16 tile
  #pragma unroll
  for (int i = 0; i < 2; ++i) {
    #pragma unroll
    for (int j = 0; j < 2; ++j) {
      int gn = n0 + wn * 32 + j * 16 + lo;
      float bv = bias[gn];
      #pragma unroll
      for (int r = 0; r < 8; ++r) {
        int grow = m0 + wm * 32 + i * 16 + r + 8 * hi;
        float vl = acc[i][j][r] + bv;
        if (EPI == 1) {
          float u = vl;
          vl = 0.5f * u * (1.0f + tanhf(0.7978845608028654f * (u + 0.044715f * u * u * u)));
        } else if (EPI == 2) {
          vl += R[(size_t)grow * N + gn];
        }
        C[(size_t)grow * N + gn] = vl;
      }
    }
  }
}

// ---------------- block-sparse flash attention ----------------
// grid: (NBLK, H, B), 128 threads (4 waves, each wave owns 16 query rows)
__global__ __launch_bounds__(128)
void attn_sparse_wmma(const float* __restrict__ q, const float* __restrict__ k,
                      const float* __restrict__ v, const int* __restrict__ bidx,
                      const float* __restrict__ amask, float* __restrict__ out)
{
  __shared__ bf16_t Qs[64][72];
  __shared__ bf16_t Ks[64][72];
  __shared__ bf16_t Vt[64][72];     // transposed: Vt[d][key]
  __shared__ bf16_t Ps[4][16][72];  // per-wave probs tile (16 rows x 64 keys)

  const int tid  = threadIdx.x;
  const int lane = tid & 31;
  const int wave = tid >> 5;
  const int lo   = lane & 15;
  const int hi   = lane >> 4;
  const int n = blockIdx.x, h = blockIdx.y, b = blockIdx.z;

  const size_t base_q = ((size_t)b * S_ + (size_t)n * BLK_) * D_ + (size_t)h * HD_;
  #pragma unroll 4
  for (int i = 0; i < 32; ++i) {
    int idx = i * 128 + tid;
    int r = idx >> 6, c = idx & 63;
    Qs[r][c] = (bf16_t)q[base_q + (size_t)r * D_ + c];
  }

  v8f oacc[4];
  #pragma unroll
  for (int t = 0; t < 4; ++t) oacc[t] = vzero8();
  float mrun[8], lrun[8];
  #pragma unroll
  for (int r = 0; r < 8; ++r) { mrun[r] = -3.0e38f; lrun[r] = 0.0f; }

  for (int jb = 0; jb < NSEL_; ++jb) {
    const int kb = bidx[n * NSEL_ + jb];
    // prefetch next gathered K/V block while this one is processed
    if (jb + 1 < NSEL_) {
      const int kb2 = bidx[n * NSEL_ + jb + 1];
      const size_t base2 = ((size_t)b * S_ + (size_t)kb2 * BLK_) * D_ + (size_t)h * HD_;
      __builtin_prefetch(&k[base2 + (size_t)(tid >> 1) * D_], 0, 1);
      __builtin_prefetch(&v[base2 + (size_t)(tid >> 1) * D_], 0, 1);
    }
    __syncthreads();  // previous iteration done reading Ks/Vt
    const size_t base_k = ((size_t)b * S_ + (size_t)kb * BLK_) * D_ + (size_t)h * HD_;
    #pragma unroll 4
    for (int i = 0; i < 32; ++i) {
      int idx = i * 128 + tid;
      int r = idx >> 6, c = idx & 63;
      float kv = k[base_k + (size_t)r * D_ + c];
      float vv = v[base_k + (size_t)r * D_ + c];
      Ks[r][c] = (bf16_t)kv;
      Vt[c][r] = (bf16_t)vv;
    }
    __syncthreads();

    // s = q @ k^T  (16x64 per wave)
    v8f sc[4];
    #pragma unroll
    for (int t = 0; t < 4; ++t) sc[t] = vzero8();
    #pragma unroll
    for (int ks = 0; ks < 2; ++ks) {
      v16bf a = ld_fragA(&Qs[wave * 16 + lo][ks * 32], hi);
      #pragma unroll
      for (int t = 0; t < 4; ++t) {
        v16bf bb = ld_fragB(&Ks[t * 16 + lo][ks * 32], hi);
        sc[t] = wmma_bf16(a, bb, sc[t]);
      }
    }

    // scale + additive padding-mask bias; row max
    float rm[8];
    #pragma unroll
    for (int r = 0; r < 8; ++r) rm[r] = -3.0e38f;
    #pragma unroll
    for (int t = 0; t < 4; ++t) {
      float am = amask[(size_t)b * S_ + (size_t)kb * BLK_ + t * 16 + lo];
      float bias = (1.0f - am) * -1.0e9f;
      #pragma unroll
      for (int r = 0; r < 8; ++r) {
        float xv = sc[t][r] * SCALE_ + bias;
        sc[t][r] = xv;
        rm[r] = fmaxf(rm[r], xv);
      }
    }
    #pragma unroll
    for (int r = 0; r < 8; ++r) {
      rm[r] = fmaxf(rm[r], __shfl_xor(rm[r], 1, 32));
      rm[r] = fmaxf(rm[r], __shfl_xor(rm[r], 2, 32));
      rm[r] = fmaxf(rm[r], __shfl_xor(rm[r], 4, 32));
      rm[r] = fmaxf(rm[r], __shfl_xor(rm[r], 8, 32));
    }

    // online softmax update
    float rs[8];
    #pragma unroll
    for (int r = 0; r < 8; ++r) {
      float mnew = fmaxf(mrun[r], rm[r]);
      float corr = __expf(mrun[r] - mnew);
      mrun[r] = mnew;
      lrun[r] *= corr;
      #pragma unroll
      for (int t = 0; t < 4; ++t) oacc[t][r] *= corr;
      rs[r] = 0.0f;
    }
    #pragma unroll
    for (int t = 0; t < 4; ++t) {
      #pragma unroll
      for (int r = 0; r < 8; ++r) {
        float p = __expf(sc[t][r] - mrun[r]);
        rs[r] += p;
        Ps[wave][r + 8 * hi][t * 16 + lo] = (bf16_t)p;
      }
    }
    #pragma unroll
    for (int r = 0; r < 8; ++r) {
      rs[r] += __shfl_xor(rs[r], 1, 32);
      rs[r] += __shfl_xor(rs[r], 2, 32);
      rs[r] += __shfl_xor(rs[r], 4, 32);
      rs[r] += __shfl_xor(rs[r], 8, 32);
      lrun[r] += rs[r];
    }
    __syncthreads();  // Ps visible for fragment reload

    // o += p @ v
    #pragma unroll
    for (int ks = 0; ks < 2; ++ks) {
      v16bf a = ld_fragA(&Ps[wave][lo][ks * 32], hi);
      #pragma unroll
      for (int t = 0; t < 4; ++t) {
        v16bf bb = ld_fragB(&Vt[t * 16 + lo][ks * 32], hi);
        oacc[t] = wmma_bf16(a, bb, oacc[t]);
      }
    }
  }

  // normalize and write [B,S,H*HD]
  #pragma unroll
  for (int t = 0; t < 4; ++t) {
    #pragma unroll
    for (int r = 0; r < 8; ++r) {
      int row = n * BLK_ + wave * 16 + r + 8 * hi;
      int d   = t * 16 + lo;
      out[((size_t)b * S_ + row) * D_ + (size_t)h * HD_ + d] = oacc[t][r] / lrun[r];
    }
  }
}

// ---------------- LayerNorm (one block per row of 768) ----------------
__device__ __forceinline__ float block_reduce_sum(float val, float* red) {
  int tid = threadIdx.x;
  red[tid] = val;
  __syncthreads();
  for (int st = 128; st > 0; st >>= 1) {
    if (tid < st) red[tid] += red[tid + st];
    __syncthreads();
  }
  float r = red[0];
  __syncthreads();
  return r;
}

__global__ __launch_bounds__(256)
void ln_rows(const float* __restrict__ in, const float* __restrict__ g,
             const float* __restrict__ bt, float* __restrict__ out)
{
  __shared__ float red[256];
  const size_t row = blockIdx.x;
  const int tid = threadIdx.x;
  const float* p = in + row * D_;
  float v0 = p[tid], v1 = p[tid + 256], v2 = p[tid + 512];
  float mu = block_reduce_sum(v0 + v1 + v2, red) * (1.0f / D_);
  float d0 = v0 - mu, d1 = v1 - mu, d2 = v2 - mu;
  float var = block_reduce_sum(d0 * d0 + d1 * d1 + d2 * d2, red) * (1.0f / D_);
  float rs = rsqrtf(var + 1e-12f);
  float* o = out + row * D_;
  o[tid]       = d0 * rs * g[tid]       + bt[tid];
  o[tid + 256] = d1 * rs * g[tid + 256] + bt[tid + 256];
  o[tid + 512] = d2 * rs * g[tid + 512] + bt[tid + 512];
}

// ---------------- embedding gather + LayerNorm ----------------
__global__ __launch_bounds__(256)
void embed_ln(const int* __restrict__ ids, const float* __restrict__ ew,
              const float* __restrict__ ep, const float* __restrict__ g,
              const float* __restrict__ bt, float* __restrict__ out)
{
  __shared__ float red[256];
  const size_t tok = blockIdx.x;
  const int tid = threadIdx.x;
  const int spos = (int)(tok % S_);
  const size_t wid = (size_t)ids[tok] * D_;
  const size_t pid = (size_t)spos * D_;
  float v0 = ew[wid + tid]       + ep[pid + tid];
  float v1 = ew[wid + tid + 256] + ep[pid + tid + 256];
  float v2 = ew[wid + tid + 512] + ep[pid + tid + 512];
  float mu = block_reduce_sum(v0 + v1 + v2, red) * (1.0f / D_);
  float d0 = v0 - mu, d1 = v1 - mu, d2 = v2 - mu;
  float var = block_reduce_sum(d0 * d0 + d1 * d1 + d2 * d2, red) * (1.0f / D_);
  float rs = rsqrtf(var + 1e-12f);
  float* o = out + tok * D_;
  o[tid]       = d0 * rs * g[tid]       + bt[tid];
  o[tid + 256] = d1 * rs * g[tid + 256] + bt[tid + 256];
  o[tid + 512] = d2 * rs * g[tid + 512] + bt[tid + 512];
}

// ---------------- pooler (tanh) + classifier tail (tiny) ----------------
__global__ __launch_bounds__(256)
void pool_cls(const float* __restrict__ x, const float* __restrict__ pw,
              const float* __restrict__ pb, const float* __restrict__ cw,
              const float* __restrict__ cb, float* __restrict__ out)
{
  __shared__ float pooled[B_ * D_];
  const int tid = threadIdx.x;
  for (int idx = tid; idx < B_ * D_; idx += 256) {
    int b = idx / D_, c = idx % D_;
    const float* xr = x + (size_t)b * S_ * D_;   // [CLS] token
    float acc = pb[c];
    for (int kk = 0; kk < D_; ++kk) acc += xr[kk] * pw[(size_t)kk * D_ + c];
    pooled[idx] = tanhf(acc);
  }
  __syncthreads();
  for (int idx = tid; idx < B_ * NLAB_; idx += 256) {
    int b = idx / NLAB_, c = idx % NLAB_;
    float acc = cb[c];
    for (int kk = 0; kk < D_; ++kk) acc += pooled[b * D_ + kk] * cw[(size_t)kk * NLAB_ + c];
    out[idx] = acc;
  }
}

// ---------------- host orchestration ----------------
extern "C" void kernel_launch(void* const* d_in, const int* in_sizes, int n_in,
                              void* d_out, int out_size, void* d_ws, size_t ws_size,
                              hipStream_t stream)
{
  (void)in_sizes; (void)n_in; (void)out_size; (void)ws_size;
  const int*   ids    = (const int*)d_in[0];
  const float* amask  = (const float*)d_in[1];
  const int*   bidx   = (const int*)d_in[2];
  const float* ew     = (const float*)d_in[3];
  const float* ep     = (const float*)d_in[4];
  const float* ln_e_g = (const float*)d_in[5];
  const float* ln_e_b = (const float*)d_in[6];
  const float* Wq = (const float*)d_in[7];   const float* bq = (const float*)d_in[8];
  const float* Wk = (const float*)d_in[9];   const float* bk = (const float*)d_in[10];
  const float* Wv = (const float*)d_in[11];  const float* bv = (const float*)d_in[12];
  const float* Wo = (const float*)d_in[13];  const float* bo = (const float*)d_in[14];
  const float* ln1_g = (const float*)d_in[15]; const float* ln1_b = (const float*)d_in[16];
  const float* W1 = (const float*)d_in[17];  const float* b1 = (const float*)d_in[18];
  const float* W2 = (const float*)d_in[19];  const float* b2 = (const float*)d_in[20];
  const float* ln2_g = (const float*)d_in[21]; const float* ln2_b = (const float*)d_in[22];
  const float* pw = (const float*)d_in[23];  const float* pb = (const float*)d_in[24];
  const float* cw = (const float*)d_in[25];  const float* cb = (const float*)d_in[26];
  float* outp = (float*)d_out;

  const size_t BSD = (size_t)B_ * S_ * D_;
  float* ws = (float*)d_ws;
  float* x  = ws;             // [B,S,D] current activations
  float* y  = x  + BSD;       // [B,S,D] GEMM scratch (pre-LN)
  float* qb = y  + BSD;       // [B,S,D]
  float* kb = qb + BSD;
  float* vb = kb + BSD;
  float* ab = vb + BSD;       // attention output
  float* hb = ab + BSD;       // [B,S,FFN] intermediate

  const int Mrows = B_ * S_;
  const dim3 blk256(256);
  const dim3 gD(D_ / BN,   Mrows / BM);   // 12 x 64
  const dim3 gF(FFN_ / BN, Mrows / BM);   // 48 x 64

  embed_ln<<<dim3(B_ * S_), blk256, 0, stream>>>(ids, ew, ep, ln_e_g, ln_e_b, x);

  for (int l = 0; l < L_; ++l) {
    const float* wq = Wq + (size_t)l * D_ * D_;
    const float* wk = Wk + (size_t)l * D_ * D_;
    const float* wv = Wv + (size_t)l * D_ * D_;
    const float* wo = Wo + (size_t)l * D_ * D_;
    const float* w1 = W1 + (size_t)l * D_ * FFN_;
    const float* w2 = W2 + (size_t)l * FFN_ * D_;

    gemm_bf16_wmma<0><<<gD, blk256, 0, stream>>>(x, wq, bq + (size_t)l * D_, nullptr, qb, Mrows, D_, D_);
    gemm_bf16_wmma<0><<<gD, blk256, 0, stream>>>(x, wk, bk + (size_t)l * D_, nullptr, kb, Mrows, D_, D_);
    gemm_bf16_wmma<0><<<gD, blk256, 0, stream>>>(x, wv, bv + (size_t)l * D_, nullptr, vb, Mrows, D_, D_);

    attn_sparse_wmma<<<dim3(NBLK_, H_, B_), dim3(128), 0, stream>>>(qb, kb, vb, bidx, amask, ab);

    // y = x + ab @ Wo + bo ; x = LN1(y)
    gemm_bf16_wmma<2><<<gD, blk256, 0, stream>>>(ab, wo, bo + (size_t)l * D_, x, y, Mrows, D_, D_);
    ln_rows<<<dim3(Mrows), blk256, 0, stream>>>(y, ln1_g + (size_t)l * D_, ln1_b + (size_t)l * D_, x);

    // h = gelu_new(x @ W1 + b1) ; y = x + h @ W2 + b2 ; x = LN2(y)
    gemm_bf16_wmma<1><<<gF, blk256, 0, stream>>>(x, w1, b1 + (size_t)l * FFN_, nullptr, hb, Mrows, FFN_, D_);
    gemm_bf16_wmma<2><<<gD, blk256, 0, stream>>>(hb, w2, b2 + (size_t)l * D_, x, y, Mrows, D_, FFN_);
    ln_rows<<<dim3(Mrows), blk256, 0, stream>>>(y, ln2_g + (size_t)l * D_, ln2_b + (size_t)l * D_, x);
  }

  pool_cls<<<dim3(1), blk256, 0, stream>>>(x, pw, pb, cw, cb, outp);
}
